// Transformer_21337397527403
// MI455X (gfx1250) — matrix operand order repaired and checked by reading, not compiled, not run on hardware
//
#include <hip/hip_runtime.h>

typedef unsigned short u16;
typedef __attribute__((ext_vector_type(16))) __bf16    v16bf;
typedef __attribute__((ext_vector_type(8)))  float     v8f;
typedef __attribute__((ext_vector_type(4)))  unsigned  v4u;
typedef __attribute__((ext_vector_type(8)))  unsigned  v8u;

#define DEV __device__ __forceinline__

// Toggle for the CDNA5 async global->LDS DMA path (inline asm).
#ifndef USE_ASYNC_LDS
#define USE_ASYNC_LDS 1
#endif

DEV u16 f2b(float f) {
  union { float f; unsigned u; } x; x.f = f;
  unsigned r = x.u + 0x7FFFu + ((x.u >> 16) & 1u);      // round-to-nearest-even
  if ((x.u & 0x7F800000u) == 0x7F800000u) r = x.u;      // inf/NaN passthrough
  return (u16)(r >> 16);
}

// 16-byte global -> LDS copy. Async variant uses CDNA5 GLOBAL_LOAD_ASYNC_TO_LDS
// (ASYNCcnt-tracked; LDS vdst = low 32 bits of the flat shared address, which is
// the wave-relative LDS byte offset per ISA 10.2 aperture rules).
DEV void g2l16(const u16* g, u16* l) {
#if USE_ASYNC_LDS
  unsigned           lo = (unsigned)(unsigned long long)l;
  unsigned long long ga = (unsigned long long)g;
  asm volatile("global_load_async_to_lds_b128 %0, %1, off" :: "v"(lo), "v"(ga) : "memory");
#else
  *(v4u*)l = *(const v4u*)g;
#endif
}
DEV void g2l_wait() {
#if USE_ASYNC_LDS
  asm volatile("s_wait_asynccnt 0x0" ::: "memory");
#endif
}

// ---------------------------------------------------------------------------
// Tiled bf16 WMMA GEMM:  C = act(A @ B + bias [+ res])
//   A: M x K row-major bf16 (lda), two-level batch strides (asO, asI)
//   B: BT==0 -> K x N row-major (ldb); BT==1 -> N x K row-major (B^T, ldb)
//   C: fp32 (Cf) and/or bf16 (Cb), shared strides csO/csI/ldc; res uses same.
//   ACT: 0 = none, 1 = exact GELU
// Block tile 128x64, 8 wave32 waves, each wave a 32x32 tile, K step = 32.
// Double-buffered LDS fed by async global->LDS DMA; branch-free clamped staging
// (K must be a multiple of 32; out-of-range rows/cols duplicate the last valid
// one and their outputs are discarded by the guarded store).
// ---------------------------------------------------------------------------
template<int BT, int ACT>
__global__ __launch_bounds__(256) void k_gemm(
    const u16* __restrict__ A, long asO, long asI, int lda,
    const u16* __restrict__ Bm, long bsO, long bsI, int ldb,
    float* __restrict__ Cf, u16* __restrict__ Cb, long csO, long csI, int ldc,
    const float* __restrict__ bias, const float* __restrict__ res,
    int M, int N, int K, int innerB)
{
  constexpr int BMt = 128, BNt = 64, KT = 32;
  __shared__ __attribute__((aligned(16))) u16 sA[2][BMt * KT];  // [m][k]
  __shared__ __attribute__((aligned(16))) u16 sB[2][BNt * KT];  // [n][k]

  const int z  = blockIdx.z;
  const int zo = z / innerB, zi = z % innerB;
  A  += (long)zo * asO + (long)zi * asI;
  Bm += (long)zo * bsO + (long)zi * bsI;
  const long coff = (long)zo * csO + (long)zi * csI;

  const int m0 = blockIdx.y * BMt;
  const int n0 = blockIdx.x * BNt;
  const int t    = threadIdx.x;
  const int wave = t >> 5;
  const int lane = t & 31;
  const int wy = wave >> 1, wx = wave & 1;   // wave tile origin (wy*32, wx*32)
  const int lh = lane >> 4, lm = lane & 15;

  auto stage = [&](int k0, int buf) {
    // A tile: 128x32 = 512 16B chunks, 2 per thread (clamped rows)
#pragma unroll
    for (int i = 0; i < 2; ++i) {
      int cidx = t + i * 256;
      int row  = cidx >> 2;
      int kc   = (cidx & 3) * 8;
      int gm = m0 + row; if (gm >= M) gm = M - 1;
      g2l16(A + (long)gm * lda + k0 + kc, &sA[buf][row * KT + kc]);
    }
    if (BT) {
      // B^T tile: 64x32 = 256 16B chunks, 1 per thread (clamped cols)
      int nn = t >> 2, kc = (t & 3) * 8;
      int gn = n0 + nn; if (gn >= N) gn = N - 1;
      g2l16(Bm + (long)gn * ldb + k0 + kc, &sB[buf][nn * KT + kc]);
    } else {
      // K x N row-major: transpose-gather into [n][k]
      int nn = t & 63, kb = (t >> 6) * 8;
      int gn = n0 + nn; if (gn >= N) gn = N - 1;
#pragma unroll
      for (int j = 0; j < 8; ++j)
        sB[buf][nn * KT + kb + j] = Bm[(long)(k0 + kb + j) * ldb + gn];
    }
  };

  v8f acc[2][2];
#pragma unroll
  for (int i = 0; i < 2; ++i)
#pragma unroll
    for (int j = 0; j < 2; ++j)
#pragma unroll
      for (int e = 0; e < 8; ++e) acc[i][j][e] = 0.0f;

  stage(0, 0);
  int buf = 0;
  for (int k0 = 0; k0 < K; k0 += KT) {
    g2l_wait();          // own wave's async copies done
    __syncthreads();     // everyone's copies visible
    if (k0 + KT < K) stage(k0 + KT, buf ^ 1);

    // ---- WMMA on current buffer (contiguous k = 16*lh + s fragments, same order for A and B)
#pragma unroll
    for (int sm = 0; sm < 2; ++sm) {
      const u16* pa = &sA[buf][(wy * 32 + sm * 16 + lm) * KT];
      v4u a0 = *(const v4u*)(pa + lh * 16);       // K {0..7} / {16..23}
      v4u a1 = *(const v4u*)(pa + lh * 16 + 8);   // K {8..15} / {24..31}
      v8u au = __builtin_shufflevector(a0, a1, 0, 1, 2, 3, 4, 5, 6, 7);
      v16bf af = __builtin_bit_cast(v16bf, au);
#pragma unroll
      for (int sn = 0; sn < 2; ++sn) {
        const u16* pb = &sB[buf][(wx * 32 + sn * 16 + lm) * KT + lh * 16];
        v4u b0 = *(const v4u*)(pb);
        v4u b1 = *(const v4u*)(pb + 8);
        v8u bu = __builtin_shufflevector(b0, b1, 0, 1, 2, 3, 4, 5, 6, 7);
        v16bf bfv = __builtin_bit_cast(v16bf, bu);
        acc[sm][sn] = __builtin_amdgcn_wmma_f32_16x16x32_bf16(
            false, af, false, bfv, (short)0, acc[sm][sn], false, false);
      }
    }
    __syncthreads();
    buf ^= 1;
  }

  // ---- epilogue: bias + residual + activation, dual fp32/bf16 store
#pragma unroll
  for (int sm = 0; sm < 2; ++sm) {
#pragma unroll
    for (int sn = 0; sn < 2; ++sn) {
      int col = n0 + wx * 32 + sn * 16 + lm;
      if (col < N) {
        float bv = bias ? bias[col] : 0.0f;
#pragma unroll
        for (int j = 0; j < 8; ++j) {
          int rowm = m0 + wy * 32 + sm * 16 + 8 * lh + j;
          if (rowm < M) {
            long o = coff + (long)rowm * ldc + col;
            float v = acc[sm][sn][j] + bv;
            if (res) v += res[o];
            if (ACT == 1) v = 0.5f * v * (1.0f + erff(v * 0.70710678118654752f));
            if (Cf) Cf[o] = v;
            if (Cb) Cb[o] = f2b(v);
          }
        }
      }
    }
  }
}

// ---------------------------------------------------------------------------
// Support kernels
// ---------------------------------------------------------------------------
// fp32 (Kd x Nd, row-major) -> bf16 transposed (Nd x Kd): every weight becomes
// B^T so GEMM B-tiles are contiguous 16B chunks (async-LDS capable).
__global__ void k_f2bt(const float* __restrict__ in, u16* __restrict__ out,
                       long Kd, long Nd) {
  long total = Kd * Nd;
  for (long i = blockIdx.x * 256l + threadIdx.x; i < total; i += (long)gridDim.x * 256) {
    long n = i / Kd, k = i - n * Kd;
    out[i] = f2b(in[k * Nd + n]);
  }
}

__global__ void k_maxpool2(const float* __restrict__ in, float* __restrict__ of,
                           u16* __restrict__ ob, long n) {
  for (long i = blockIdx.x * 256l + threadIdx.x; i < n; i += (long)gridDim.x * 256) {
    float v = fmaxf(in[2 * i], in[2 * i + 1]);
    of[i] = v;
    if (ob) ob[i] = f2b(v);
  }
}

template<typename T>
__global__ void k_copy2d(const T* __restrict__ src, long ss, T* __restrict__ dst,
                         long ds, long rowlen, long rows) {
  long total = rows * rowlen;
  for (long i = blockIdx.x * 256l + threadIdx.x; i < total; i += (long)gridDim.x * 256) {
    long r = i / rowlen, c = i - r * rowlen;
    dst[r * ds + c] = src[r * ss + c];
  }
}

__global__ __launch_bounds__(256) void k_layernorm(
    const float* __restrict__ x, const float* __restrict__ g,
    const float* __restrict__ b, u16* __restrict__ out, int n)
{
  int row = blockIdx.x;
  const float* xr = x + (long)row * n;
  u16* orow = out + (long)row * n;
  float s = 0.0f, ss = 0.0f;
  for (int i = threadIdx.x; i < n; i += 256) { float v = xr[i]; s += v; ss += v * v; }
  __shared__ float r0[256], r1[256];
  r0[threadIdx.x] = s; r1[threadIdx.x] = ss;
  __syncthreads();
  for (int o = 128; o > 0; o >>= 1) {
    if (threadIdx.x < o) { r0[threadIdx.x] += r0[threadIdx.x + o]; r1[threadIdx.x] += r1[threadIdx.x + o]; }
    __syncthreads();
  }
  float mean = r0[0] / n;
  float var  = r1[0] / n - mean * mean;
  float inv  = rsqrtf(var + 1e-5f);
  for (int i = threadIdx.x; i < n; i += 256)
    orow[i] = f2b((xr[i] - mean) * inv * g[i] + b[i]);
}

__global__ __launch_bounds__(256) void k_softmax(
    const float* __restrict__ s, u16* __restrict__ out, int n, float scale, long rows)
{
  long row = blockIdx.x * 8l + (threadIdx.x >> 5);
  if (row >= rows) return;
  int lane = threadIdx.x & 31;
  const float* sr = s + row * n;
  u16* orow = out + row * n;
  float mx = -1e30f;
  for (int i = lane; i < n; i += 32) mx = fmaxf(mx, sr[i] * scale);
  for (int o = 16; o > 0; o >>= 1) mx = fmaxf(mx, __shfl_xor(mx, o, 32));
  float sum = 0.0f;
  for (int i = lane; i < n; i += 32) sum += expf(sr[i] * scale - mx);
  for (int o = 16; o > 0; o >>= 1) sum += __shfl_xor(sum, o, 32);
  float inv = 1.0f / sum;
  for (int i = lane; i < n; i += 32) orow[i] = f2b(expf(sr[i] * scale - mx) * inv);
}

__global__ __launch_bounds__(256) void k_conv1d(
    const float* __restrict__ x, const float* __restrict__ w,
    const float* __restrict__ bias, float* __restrict__ y, int L)
{
  int o = blockIdx.y, b = blockIdx.z;
  __shared__ float sw[64 * 7];
  for (int i = threadIdx.x; i < 448; i += 256) sw[i] = w[o * 448 + i];
  __syncthreads();
  int l = blockIdx.x * 256 + threadIdx.x;
  if (l >= L) return;
  const float* xb = x + (long)b * 64 * L;
  float acc = bias[o];
  for (int ci = 0; ci < 64; ++ci) {
    const float* xr = xb + (long)ci * L;
#pragma unroll
    for (int k = 0; k < 7; ++k) {
      int p = l + k - 3;
      float xv = (p >= 0 && p < L) ? xr[p] : 0.0f;
      acc += sw[ci * 7 + k] * xv;
    }
  }
  y[((long)b * 64 + o) * L + l] = acc;
}

__global__ __launch_bounds__(256) void k_bnstats(
    const float* __restrict__ y, float* __restrict__ stats, int Bn, int L)
{
  int ch = blockIdx.x;
  long n = (long)Bn * L;
  float s = 0.0f, ss = 0.0f;
  for (long i = threadIdx.x; i < n; i += 256) {
    long b = i / L, l = i - b * L;
    float v = y[(b * 64 + ch) * L + l];
    s += v; ss += v * v;
  }
  __shared__ float r0[256], r1[256];
  r0[threadIdx.x] = s; r1[threadIdx.x] = ss;
  __syncthreads();
  for (int o = 128; o > 0; o >>= 1) {
    if (threadIdx.x < o) { r0[threadIdx.x] += r0[threadIdx.x + o]; r1[threadIdx.x] += r1[threadIdx.x + o]; }
    __syncthreads();
  }
  if (threadIdx.x == 0) {
    float m = r0[0] / (float)n;
    stats[ch] = m;
    stats[64 + ch] = r1[0] / (float)n - m * m;
  }
}

__global__ void k_bnep(const float* __restrict__ y, const float* __restrict__ stats,
                       const float* __restrict__ g, const float* __restrict__ b,
                       float* __restrict__ out, int Bn, int L)
{
  int half = L >> 1;
  long total = (long)Bn * 64 * half;
  for (long i = blockIdx.x * 256l + threadIdx.x; i < total; i += (long)gridDim.x * 256) {
    long bc = i / half;
    int l = (int)(i - bc * half);
    int ch = (int)(bc & 63);
    float m = stats[ch], inv = rsqrtf(stats[64 + ch] + 1e-5f);
    float gg = g[ch], bb = b[ch];
    const float* yr = y + bc * L;
    float a = (yr[2 * l] - m) * inv * gg + bb;
    float c = (yr[2 * l + 1] - m) * inv * gg + bb;
    a = a > 0.0f ? a : expm1f(a);
    c = c > 0.0f ? c : expm1f(c);
    out[i] = fmaxf(a, c);
  }
}

__global__ __launch_bounds__(64) void k_lms(const float* __restrict__ x,
                                            float* __restrict__ out, int len)
{
  int row = blockIdx.x;
  const float* xr = x + (long)row * len;
  float s = 0.0f;
  for (int i = threadIdx.x; i < len; i += 64) { float v = xr[i]; s += v * v; }
  __shared__ float red[64];
  red[threadIdx.x] = s;
  __syncthreads();
  for (int o = 32; o > 0; o >>= 1) {
    if (threadIdx.x < o) red[threadIdx.x] += red[threadIdx.x + o];
    __syncthreads();
  }
  if (threadIdx.x == 0) out[row] = logf(red[0] / (float)len);
}

// ---------------------------------------------------------------------------
// Host side
// ---------------------------------------------------------------------------
static inline unsigned nblk(long n) {
  long g = (n + 255) / 256;
  if (g < 1) g = 1;
  if (g > 8192) g = 8192;
  return (unsigned)g;
}

static void launch_gemm(hipStream_t st, int BT, int ACT,
    const u16* A, long asO, long asI, int lda,
    const u16* Bm, long bsO, long bsI, int ldb,
    float* Cf, u16* Cb, long csO, long csI, int ldc,
    const float* bias, const float* res,
    int M, int N, int K, int batch, int innerB)
{
  dim3 grid((unsigned)((N + 63) / 64), (unsigned)((M + 127) / 128), (unsigned)batch);
  dim3 blk(256, 1, 1);
  if (BT && ACT) {
    k_gemm<1, 1><<<grid, blk, 0, st>>>(A, asO, asI, lda, Bm, bsO, bsI, ldb,
                                       Cf, Cb, csO, csI, ldc, bias, res, M, N, K, innerB);
  } else if (BT) {
    k_gemm<1, 0><<<grid, blk, 0, st>>>(A, asO, asI, lda, Bm, bsO, bsI, ldb,
                                       Cf, Cb, csO, csI, ldc, bias, res, M, N, K, innerB);
  } else if (ACT) {
    k_gemm<0, 1><<<grid, blk, 0, st>>>(A, asO, asI, lda, Bm, bsO, bsI, ldb,
                                       Cf, Cb, csO, csI, ldc, bias, res, M, N, K, innerB);
  } else {
    k_gemm<0, 0><<<grid, blk, 0, st>>>(A, asO, asI, lda, Bm, bsO, bsI, ldb,
                                       Cf, Cb, csO, csI, ldc, bias, res, M, N, K, innerB);
  }
}

enum {
  WQ = 0, WK, WV, WO_, BO_, FF_LG, FF_LB, FF_W1, FF_B1, FF_W2, FF_B2,
  CW, CB_, BNG, BNB, QW, QB, KW, KB, VW, VB,
  XWO, XBO, XLG, XLB, XW1, XB1, XW2, XB2
};

extern "C" void kernel_launch(void* const* d_in, const int* in_sizes, int n_in,
                              void* d_out, int out_size, void* d_ws, size_t ws_size,
                              hipStream_t stream)
{
  (void)in_sizes; (void)n_in; (void)out_size; (void)ws_size;

  const int Bz = 128, SEQ = 64, Mrows = Bz * SEQ;  // 8192 rows

  auto PP = [&](int d, int c, int idx) -> const float* {
    return (const float*)d_in[4 + (d * 4 + c) * 29 + idx];
  };
  auto MC = [&](int c, int idx) -> const float* {
    return (const float*)d_in[4 + 232 + c * 6 + idx];
  };
  auto OO = [&](int idx) -> const float* {
    return (const float*)d_in[4 + 232 + 24 + idx];
  };

  // ---- bump allocator over workspace
  char* base = (char*)d_ws;
  size_t off = 0;
  auto alloc = [&](size_t bytes) -> void* {
    size_t cur = (off + 255) & ~(size_t)255;
    off = cur + bytes;
    return (void*)(base + cur);
  };

  // persistent buffers
  float* chanf[4];
  for (int c = 0; c < 4; ++c) chanf[c] = (float*)alloc((size_t)Mrows * 512 * 4);
  float* densef = (float*)alloc((size_t)2 * 4 * Mrows * 4);
  u16 *qsb[4], *ksb[4], *vsb[4], *xsb[4];
  for (int c = 0; c < 4; ++c) {
    qsb[c] = (u16*)alloc((size_t)Mrows * 512 * 2);
    ksb[c] = (u16*)alloc((size_t)Mrows * 512 * 2);
    vsb[c] = (u16*)alloc((size_t)Mrows * 512 * 2);
    xsb[c] = (u16*)alloc((size_t)Mrows * 512 * 2);
  }
  u16* kagg = (u16*)alloc((size_t)Bz * 192 * 512 * 2);
  u16* vagg = (u16*)alloc((size_t)Bz * 192 * 512 * 2);
  u16* wst  = (u16*)alloc((size_t)16512 * 128 * 2);   // transposed bf16 weight staging
  const size_t mark = off;

  // convert weight (Kd x Nd fp32) -> wst as (Nd x Kd) bf16 (= B^T, ldb = Kd)
  auto toBfT = [&](const float* w, long Kd, long Nd) {
    k_f2bt<<<dim3(nblk(Kd * Nd)), dim3(256), 0, stream>>>(w, wst, Kd, Nd);
  };

  for (int d = 0; d < 2; ++d) {
    int Lin = 1024 >> d;
    int dim = Lin >> 1;

    // ======== per-channel local branches ========
    for (int c = 0; c < 4; ++c) {
      off = mark;
      const float* xin = (d == 0) ? (const float*)d_in[c] : chanf[c];
      float* xcgf   = (float*)alloc((size_t)Mrows * dim * 4);
      u16*   xcgb   = (u16*)  alloc((size_t)Mrows * dim * 2);
      u16*   qcg    = (u16*)  alloc((size_t)Mrows * 512 * 2);
      u16*   kcg    = (u16*)  alloc((size_t)Mrows * 512 * 2);
      u16*   vcg    = (u16*)  alloc((size_t)Mrows * 512 * 2);
      float* scoresf= (float*)alloc((size_t)Bz * 8 * 64 * 64 * 4);
      u16*   attnb  = (u16*)  alloc((size_t)Bz * 8 * 64 * 64 * 2);
      u16*   ab     = (u16*)  alloc((size_t)Mrows * 512 * 2);
      float* xcg2f  = (float*)alloc((size_t)Mrows * dim * 4);
      float* yconv  = (float*)alloc((size_t)Bz * 64 * Lin * 4);
      float* stats  = (float*)alloc(128 * 4);
      float* xfgf   = (float*)alloc((size_t)Mrows * dim * 4);
      u16*   tlnb   = (u16*)  alloc((size_t)Mrows * dim * 2);
      u16*   hb     = (u16*)  alloc((size_t)Mrows * 256 * 2);

      long nP = (long)Mrows * dim;
      // coarse branch: maxpool (fp32 + bf16)
      k_maxpool2<<<dim3(nblk(nP)), 256, 0, stream>>>(xin, xcgf, xcgb, nP);
      // self-attn projections (no bias), weights pre-transposed
      toBfT(PP(d, c, WQ), dim, 512);
      launch_gemm(stream, 1, 0, xcgb, 0, 0, dim, wst, 0, 0, dim,
                  nullptr, qcg, 0, 0, 512, nullptr, nullptr, Mrows, 512, dim, 1, 1);
      toBfT(PP(d, c, WK), dim, 512);
      launch_gemm(stream, 1, 0, xcgb, 0, 0, dim, wst, 0, 0, dim,
                  nullptr, kcg, 0, 0, 512, nullptr, nullptr, Mrows, 512, dim, 1, 1);
      toBfT(PP(d, c, WV), dim, 512);
      launch_gemm(stream, 1, 0, xcgb, 0, 0, dim, wst, 0, 0, dim,
                  nullptr, vcg, 0, 0, 512, nullptr, nullptr, Mrows, 512, dim, 1, 1);
      // scores = q @ k^T  (batched over B*H, head-sliced; K already n-major)
      launch_gemm(stream, 1, 0, qcg, 64l * 512, 64, 512, kcg, 64l * 512, 64, 512,
                  scoresf, nullptr, 8l * 64 * 64, 64l * 64, 64,
                  nullptr, nullptr, 64, 64, 64, Bz * 8, 8);
      long rows1 = (long)Bz * 8 * 64;
      k_softmax<<<dim3((unsigned)((rows1 + 7) / 8)), 256, 0, stream>>>(scoresf, attnb, 64, 0.125f, rows1);
      // o = attn @ v  (v is K x N row-major slice -> gather path)
      launch_gemm(stream, 0, 0, attnb, 8l * 64 * 64, 64l * 64, 64, vcg, 64l * 512, 64, 512,
                  nullptr, ab, 64l * 512, 64, 512, nullptr, nullptr, 64, 64, 64, Bz * 8, 8);
      // x_cg = a @ wo + bo + x_cg
      toBfT(PP(d, c, WO_), 512, dim);
      launch_gemm(stream, 1, 0, ab, 0, 0, 512, wst, 0, 0, 512,
                  xcg2f, nullptr, 0, 0, dim, PP(d, c, BO_), xcgf, Mrows, dim, 512, 1, 1);
      // fine branch: conv -> BN(batch stats) -> ELU -> pool
      k_conv1d<<<dim3((unsigned)((Lin + 255) / 256), 64, Bz), 256, 0, stream>>>(
          xin, PP(d, c, CW), PP(d, c, CB_), yconv, Lin);
      k_bnstats<<<dim3(64), 256, 0, stream>>>(yconv, stats, Bz, Lin);
      k_bnep<<<dim3(nblk(nP)), 256, 0, stream>>>(yconv, stats, PP(d, c, BNG), PP(d, c, BNB), xfgf, Bz, Lin);
      // dense info: log(mean(x_fg^2))
      k_lms<<<dim3(Mrows), 64, 0, stream>>>(xfgf, densef + ((long)d * 4 + c) * Mrows, dim);
      // x = ffn(x_cg) + x_fg
      k_layernorm<<<dim3(Mrows), 256, 0, stream>>>(xcg2f, PP(d, c, FF_LG), PP(d, c, FF_LB), tlnb, dim);
      toBfT(PP(d, c, FF_W1), dim, 256);
      launch_gemm(stream, 1, 1, tlnb, 0, 0, dim, wst, 0, 0, dim,
                  nullptr, hb, 0, 0, 256, PP(d, c, FF_B1), nullptr, Mrows, 256, dim, 1, 1);
      toBfT(PP(d, c, FF_W2), 256, dim);
      launch_gemm(stream, 1, 0, hb, 0, 0, 256, wst, 0, 0, 256,
                  chanf[c], xsb[c], 0, 0, dim, PP(d, c, FF_B2), xfgf, Mrows, dim, 256, 1, 1);
      // cross-attention projections (with bias)
      toBfT(PP(d, c, QW), dim, 512);
      launch_gemm(stream, 1, 0, xsb[c], 0, 0, dim, wst, 0, 0, dim,
                  nullptr, qsb[c], 0, 0, 512, PP(d, c, QB), nullptr, Mrows, 512, dim, 1, 1);
      toBfT(PP(d, c, KW), dim, 512);
      launch_gemm(stream, 1, 0, xsb[c], 0, 0, dim, wst, 0, 0, dim,
                  nullptr, ksb[c], 0, 0, 512, PP(d, c, KB), nullptr, Mrows, 512, dim, 1, 1);
      toBfT(PP(d, c, VW), dim, 512);
      launch_gemm(stream, 1, 0, xsb[c], 0, 0, dim, wst, 0, 0, dim,
                  nullptr, vsb[c], 0, 0, 512, PP(d, c, VB), nullptr, Mrows, 512, dim, 1, 1);
    }

    // ======== cross-channel attention ========
    for (int c = 0; c < 4; ++c) {
      off = mark;
      float* scoresX = (float*)alloc((size_t)Bz * 8 * 64 * 192 * 4);
      u16*   attnX   = (u16*)  alloc((size_t)Bz * 8 * 64 * 192 * 2);
      u16*   ob      = (u16*)  alloc((size_t)Mrows * 512 * 2);
      float* x2f     = (float*)alloc((size_t)Mrows * dim * 4);
      u16*   tln2    = (u16*)  alloc((size_t)Mrows * dim * 2);
      u16*   h2      = (u16*)  alloc((size_t)Mrows * 1024 * 2);

      // concat sibling K/V along token axis
      int slot = 0;
      long rowlen = 64l * 512;
      for (int j = 0; j < 4; ++j) {
        if (j == c) continue;
        k_copy2d<u16><<<dim3(nblk(rowlen * Bz)), 256, 0, stream>>>(
            ksb[j], rowlen, kagg + (long)slot * rowlen, 192l * 512, rowlen, Bz);
        k_copy2d<u16><<<dim3(nblk(rowlen * Bz)), 256, 0, stream>>>(
            vsb[j], rowlen, vagg + (long)slot * rowlen, 192l * 512, rowlen, Bz);
        ++slot;
      }
      launch_gemm(stream, 1, 0, qsb[c], 64l * 512, 64, 512, kagg, 192l * 512, 64, 512,
                  scoresX, nullptr, 8l * 64 * 192, 64l * 192, 192,
                  nullptr, nullptr, 64, 192, 64, Bz * 8, 8);
      long rowsX = (long)Bz * 8 * 64;
      k_softmax<<<dim3((unsigned)((rowsX + 7) / 8)), 256, 0, stream>>>(scoresX, attnX, 192, 0.125f, rowsX);
      launch_gemm(stream, 0, 0, attnX, 8l * 64 * 192, 64l * 192, 192, vagg, 192l * 512, 64, 512,
                  nullptr, ob, 64l * 512, 64, 512, nullptr, nullptr, 64, 64, 192, Bz * 8, 8);
      // x = xs + o @ x_wo + x_bo
      toBfT(PP(d, c, XWO), 512, dim);
      launch_gemm(stream, 1, 0, ob, 0, 0, 512, wst, 0, 0, 512,
                  x2f, nullptr, 0, 0, dim, PP(d, c, XBO), chanf[c], Mrows, dim, 512, 1, 1);
      // x = ffn(x)   (replaces x)
      k_layernorm<<<dim3(Mrows), 256, 0, stream>>>(x2f, PP(d, c, XLG), PP(d, c, XLB), tln2, dim);
      toBfT(PP(d, c, XW1), dim, 1024);
      launch_gemm(stream, 1, 1, tln2, 0, 0, dim, wst, 0, 0, dim,
                  nullptr, h2, 0, 0, 1024, PP(d, c, XB1), nullptr, Mrows, 1024, dim, 1, 1);
      toBfT(PP(d, c, XW2), 1024, dim);
      launch_gemm(stream, 1, 0, h2, 0, 0, 1024, wst, 0, 0, 1024,
                  chanf[c], nullptr, 0, 0, dim, PP(d, c, XB2), nullptr, Mrows, dim, 1024, 1, 1);
    }
  }

  // ======== per-channel compression heads + output head ========
  {
    off = mark;
    const int CI = 16512;
    float* compf = (float*)alloc((size_t)Bz * CI * 4);
    u16*   tlnh  = (u16*)  alloc((size_t)Bz * CI * 2);
    u16*   hh    = (u16*)  alloc((size_t)Bz * 128 * 2);
    float* embf  = (float*)alloc((size_t)Bz * 512 * 4);
    u16*   tlne  = (u16*)  alloc((size_t)Bz * 512 * 2);
    u16*   he    = (u16*)  alloc((size_t)Bz * 512 * 2);

    for (int c = 0; c < 4; ++c) {
      k_copy2d<float><<<dim3(nblk((long)Bz * 16384)), 256, 0, stream>>>(
          chanf[c], 16384, compf, CI, 16384, Bz);
      k_copy2d<float><<<dim3(nblk((long)Bz * 64)), 256, 0, stream>>>(
          densef + (long)(0 * 4 + c) * Mrows, 64, compf + 16384, CI, 64, Bz);
      k_copy2d<float><<<dim3(nblk((long)Bz * 64)), 256, 0, stream>>>(
          densef + (long)(1 * 4 + c) * Mrows, 64, compf + 16448, CI, 64, Bz);
      k_layernorm<<<dim3(Bz), 256, 0, stream>>>(compf, MC(c, 0), MC(c, 1), tlnh, CI);
      toBfT(MC(c, 2), CI, 128);
      launch_gemm(stream, 1, 1, tlnh, 0, 0, CI, wst, 0, 0, CI,
                  nullptr, hh, 0, 0, 128, MC(c, 3), nullptr, Bz, 128, CI, 1, 1);
      toBfT(MC(c, 4), 128, 128);
      launch_gemm(stream, 1, 0, hh, 0, 0, 128, wst, 0, 0, 128,
                  embf + c * 128, nullptr, 0, 0, 512, MC(c, 5), nullptr, Bz, 128, 128, 1, 1);
    }
    k_layernorm<<<dim3(Bz), 256, 0, stream>>>(embf, OO(0), OO(1), tlne, 512);
    toBfT(OO(2), 512, 512);
    launch_gemm(stream, 1, 1, tlne, 0, 0, 512, wst, 0, 0, 512,
                nullptr, he, 0, 0, 512, OO(3), nullptr, Bz, 512, 512, 1, 1);
    toBfT(OO(4), 512, 64);
    launch_gemm(stream, 1, 0, he, 0, 0, 512, wst, 0, 0, 512,
                (float*)d_out, nullptr, 0, 0, 64, OO(5), nullptr, Bz, 64, 512, 1, 1);
  }
}
